// MLPPredictor_5403068859073
// MI455X (gfx1250) — matrix operand order repaired and checked by reading, not compiled, not run on hardware
//
#include <hip/hip_runtime.h>

// Sizes from the reference
#define NNODES 100000
#define NEDGES 600000
#define DFEAT  128
#define EDIM   128
#define NCLS   10
#define WROW   384          // 2*DFEAT + EDIM columns of W

typedef __attribute__((ext_vector_type(4)))  float    v4f;
typedef __attribute__((ext_vector_type(8)))  float    v8f;
typedef __attribute__((ext_vector_type(8)))  _Float16 v8h;
typedef __attribute__((ext_vector_type(16))) _Float16 v16h;

// ---- WMMA operand loaders -------------------------------------------------
// B (32x16, f16): lane L holds 16 consecutive halves K = 16*(L>>4) + 0..15 of
// weight row n = L&15. Caller passes p already offset by row/K-group.
__device__ __forceinline__ v16h load_b_tile(const _Float16* p) {
    v8h lo = *(const v8h*)(p);
    v8h hi = *(const v8h*)(p + 8);
    return __builtin_shufflevector(lo, hi, 0,1,2,3,4,5,6,7,8,9,10,11,12,13,14,15);
}

// A (16x32, f16 from f32 source): lane L (M=L&15, g=L>>4) needs
// K = kbase + 16g + [0,16)  (elements 0..15).
// Caller passes p = row + kbase + 16g (16B aligned).
__device__ __forceinline__ v16h load_a_tile(const float* p) {
    v4f c0 = *(const v4f*)(p);
    v4f c1 = *(const v4f*)(p + 4);
    v4f c2 = *(const v4f*)(p + 8);
    v4f c3 = *(const v4f*)(p + 12);
    v16h a;
#pragma unroll
    for (int i = 0; i < 4; ++i) {
        a[i]      = (_Float16)c0[i];
        a[i + 4]  = (_Float16)c1[i];
        a[i + 8]  = (_Float16)c2[i];
        a[i + 12] = (_Float16)c3[i];
    }
    return a;
}

// ---- Kernel 1: convert W to zero-padded [16][384] f16 + padded bias -------
__global__ void prep_w16(const float* __restrict__ Ww, const float* __restrict__ Wb,
                         _Float16* __restrict__ W16, float* __restrict__ bias16) {
    int idx = blockIdx.x * blockDim.x + threadIdx.x;   // 0 .. 16*384-1
    int n = idx / WROW;
    W16[idx] = (n < NCLS) ? (_Float16)Ww[idx] : (_Float16)0.0f;
    if (idx < 16) bias16[idx] = (idx < NCLS) ? Wb[idx] : 0.0f;
}

// ---- Kernel 2: per-node score tables  s_u = h@Wu^T, s_v = h@Wv^T ----------
// One wave per 16 nodes; two f32 accumulators share the A tile.
__global__ void node_scores(const float* __restrict__ h, const _Float16* __restrict__ W16,
                            float* __restrict__ su, float* __restrict__ sv) {
    int wave = threadIdx.x >> 5;
    int lane = threadIdx.x & 31;
    int tile = blockIdx.x * (blockDim.x >> 5) + wave;
    if (tile * 16 >= NNODES) return;                   // wave-uniform; EXEC stays full
    int m0 = tile * 16;
    int M = lane & 15;
    int g = lane >> 4;

    const _Float16* wrow = W16 + (size_t)M * WROW + g * 16;     // Wu cols 0..127
    const float*    arow = h   + (size_t)(m0 + M) * DFEAT + g * 16;

    v8f accu = {};
    v8f accv = {};
#pragma unroll
    for (int kk = 0; kk < 4; ++kk) {
        v16h a  = load_a_tile(arow + kk * 32);
        v16h bu = load_b_tile(wrow + kk * 32);
        v16h bv = load_b_tile(wrow + DFEAT + kk * 32);          // Wv cols 128..255
        accu = __builtin_amdgcn_wmma_f32_16x16x32_f16(false, a, false, bu,
                                                      (short)0, accu, false, false);
        accv = __builtin_amdgcn_wmma_f32_16x16x32_f16(false, a, false, bv,
                                                      (short)0, accv, false, false);
    }
    // D layout: lane L, reg r -> row r + 8g, col L&15. Padded [N][16] rows.
    int c = M;
#pragma unroll
    for (int r = 0; r < 8; ++r) {
        int node = m0 + r + g * 8;
        su[node * 16 + c] = accu[r];
        sv[node * 16 + c] = accv[r];
    }
}

// ---- Kernel 3: edge kernel  out = edge_h@We^T + s_u[src] + s_v[dst] + b ---
__global__ void edge_scores(const float* __restrict__ eh, const _Float16* __restrict__ W16,
                            const float* __restrict__ bias16,
                            const float* __restrict__ su, const float* __restrict__ sv,
                            const int* __restrict__ src, const int* __restrict__ dst,
                            float* __restrict__ out) {
    int wave = threadIdx.x >> 5;
    int lane = threadIdx.x & 31;
    int tile = blockIdx.x * (blockDim.x >> 5) + wave;  // 37500 tiles exactly
    int e0 = tile * 16;
    int M = lane & 15;
    int g = lane >> 4;

    const _Float16* wrow = W16 + (size_t)M * WROW + 2 * DFEAT + g * 16; // We cols
    const float*    arow = eh  + (size_t)(e0 + M) * EDIM + g * 16;

    // We (128x16) resident in VGPRs: 4 x v16h = 32 VGPRs
    v16h b0 = load_b_tile(wrow);
    v16h b1 = load_b_tile(wrow + 32);
    v16h b2 = load_b_tile(wrow + 64);
    v16h b3 = load_b_tile(wrow + 96);

    v8f acc = {};
    acc = __builtin_amdgcn_wmma_f32_16x16x32_f16(false, load_a_tile(arow),
                                                 false, b0, (short)0, acc, false, false);
    acc = __builtin_amdgcn_wmma_f32_16x16x32_f16(false, load_a_tile(arow + 32),
                                                 false, b1, (short)0, acc, false, false);
    acc = __builtin_amdgcn_wmma_f32_16x16x32_f16(false, load_a_tile(arow + 64),
                                                 false, b2, (short)0, acc, false, false);
    acc = __builtin_amdgcn_wmma_f32_16x16x32_f16(false, load_a_tile(arow + 96),
                                                 false, b3, (short)0, acc, false, false);

    int c = M;
    float bc = bias16[c];
#pragma unroll
    for (int r = 0; r < 8; ++r) {
        int e = e0 + r + g * 8;
        int s = src[e];
        int d = dst[e];
        float val = acc[r] + su[s * 16 + c] + sv[d * 16 + c] + bc;
        if (c < NCLS) out[(size_t)e * NCLS + c] = val;  // cols 10..15 discarded
    }
}

// ---- Host-side launch -----------------------------------------------------
extern "C" void kernel_launch(void* const* d_in, const int* in_sizes, int n_in,
                              void* d_out, int out_size, void* d_ws, size_t ws_size,
                              hipStream_t stream) {
    (void)in_sizes; (void)n_in; (void)out_size; (void)ws_size;
    const float* h   = (const float*)d_in[0];
    const float* eh  = (const float*)d_in[1];
    const float* Ww  = (const float*)d_in[2];
    const float* Wb  = (const float*)d_in[3];
    const int*   src = (const int*)d_in[4];
    const int*   dst = (const int*)d_in[5];
    float* out = (float*)d_out;

    // Workspace layout (total ~12.8 MB):
    //   [0, 12288)            W16: f16 [16][384], rows >= 10 zeroed
    //   [12288, 12352)        bias16: f32 [16], padded with zeros
    //   [16384, +6.4MB)       s_u: f32 [NNODES][16]
    //   [.., +6.4MB)          s_v: f32 [NNODES][16]
    _Float16* W16    = (_Float16*)d_ws;
    float*    bias16 = (float*)((char*)d_ws + 12288);
    float*    su     = (float*)((char*)d_ws + 16384);
    float*    sv     = su + (size_t)NNODES * 16;

    prep_w16<<<dim3((16 * WROW) / 256), dim3(256), 0, stream>>>(Ww, Wb, W16, bias16);

    int node_tiles = (NNODES + 15) / 16;               // 6250
    int node_blocks = (node_tiles + 3) / 4;            // 4 waves / block
    node_scores<<<dim3(node_blocks), dim3(128), 0, stream>>>(h, W16, su, sv);

    int edge_blocks = (NEDGES / 16) / 4;               // 37500 tiles / 4 waves = 9375
    edge_scores<<<dim3(edge_blocks), dim3(128), 0, stream>>>(eh, W16, bias16, su, sv,
                                                             src, dst, out);
}
